// CrossAttention_52862457479820
// MI455X (gfx1250) — matrix operand order repaired and checked by reading, not compiled, not run on hardware
//
#include <hip/hip_runtime.h>

typedef __bf16 bf16_t;
typedef __bf16 v16bf __attribute__((ext_vector_type(16)));
typedef __bf16 v8bf  __attribute__((ext_vector_type(8)));
typedef __bf16 v4bf  __attribute__((ext_vector_type(4)));
typedef float  v8f   __attribute__((ext_vector_type(8)));

#define BATCH 8
#define SEQ   1024
#define CDIM  768
#define NH    12
#define HD    64
#define ROWS  (BATCH*SEQ)   /* 8192 */
#define NT    (CDIM/64)     /* 12 column blocks == heads */
#define MT    (ROWS/16)     /* 512 row tiles */
#define QW    8             /* waves (q-tiles) per attention workgroup */

static __device__ inline v8f wmma_bf16(v16bf a, v16bf b, v8f c) {
  return __builtin_amdgcn_wmma_f32_16x16x32_bf16(false, a, false, b, (short)0, c,
                                                 false, false);
}

// A/B fragment: 8 halves at p[0..7] and 8 halves at p[16..23] (ISA per-lane layout)
static __device__ inline v16bf ld2x8(const bf16_t* p) {
  v8bf lo = *(const v8bf*)(p);
  v8bf hv = *(const v8bf*)(p + 16);
  return __builtin_shufflevector(lo, hv, 0,1,2,3,4,5,6,7,8,9,10,11,12,13,14,15);
}

// Async 16B global -> LDS copy (ASYNCcnt-tracked). LDS byte address is the low
// 32 bits of the generic pointer.
static __device__ inline void async_cp16(const bf16_t* g, bf16_t* l) {
  unsigned lds = (unsigned)(uintptr_t)l;
  unsigned long long ga = (unsigned long long)(uintptr_t)g;
  asm volatile("global_load_async_to_lds_b128 %0, %1, off"
               :: "v"(lds), "v"(ga) : "memory");
}
static __device__ inline void wait_async0() {
  asm volatile("s_wait_asynccnt 0x0" ::: "memory");
}

// fp32 -> bf16 streaming convert, 4 elems/thread
__global__ __launch_bounds__(256) void cvt_kernel(const float* __restrict__ s,
                                                  bf16_t* __restrict__ d, int n4) {
  int i = blockIdx.x * 256 + threadIdx.x;
  if (i < n4) {
    const float4 f = ((const float4*)s)[i];
    v4bf r;
    r[0] = (bf16_t)f.x; r[1] = (bf16_t)f.y; r[2] = (bf16_t)f.z; r[3] = (bf16_t)f.w;
    ((v4bf*)d)[i] = r;
  }
}

// MODE 0: Q/K projection + fused RoPE2D, writes bf16 [b][h][n][hd]
// MODE 1: V projection, writes bf16 transposed [b][h][hd][n]
// MODE 2: output projection + bias, writes fp32 [b][n][c]
template <int MODE>
__global__ __launch_bounds__(32) void proj_kernel(
    const bf16_t* __restrict__ Ab, const bf16_t* __restrict__ Wb,
    const int* __restrict__ pos, const float* __restrict__ bias,
    bf16_t* __restrict__ outb, float* __restrict__ outf) {
  const int tile = blockIdx.x;
  const int mt = tile / NT;
  const int nt = tile % NT;
  const int lane = threadIdx.x & 31;
  const int lmod = lane & 15;
  const int hi = lane >> 4;

  const int arow = mt * 16 + lmod;
  const int kbA = hi ? 16 : 0;

  v8f acc0 = {}, acc1 = {}, acc2 = {}, acc3 = {};

  for (int kc = 0; kc < CDIM / 32; ++kc) {
    v16bf a = *(const v16bf*)(Ab + (size_t)arow * CDIM + kc * 32 + kbA);
    // B[k=c, n=d] = W[d][c]; lane's column d = nt*64 + t*16 + lmod
    const bf16_t* wb = Wb + (size_t)(nt * 64 + lmod) * CDIM + kc * 32 + (hi ? 16 : 0);
    acc0 = wmma_bf16(a, *(const v16bf*)(wb + (size_t)0 * CDIM), acc0);
    acc1 = wmma_bf16(a, *(const v16bf*)(wb + (size_t)16 * CDIM), acc1);
    acc2 = wmma_bf16(a, *(const v16bf*)(wb + (size_t)32 * CDIM), acc2);
    acc3 = wmma_bf16(a, *(const v16bf*)(wb + (size_t)48 * CDIM), acc3);
  }

  if (MODE == 0) {
    // RoPE: head dims [0..31]=y (pos[...,0]), [32..63]=x (pos[...,1]);
    // pair (i, i+16) rotates by angle pos * 100^(-i/16); i = lmod.
    const float fi = __expf(-0.28782313662425572f * (float)lmod);
#pragma unroll
    for (int v = 0; v < 8; ++v) {
      int g = mt * 16 + v + 8 * hi;  // global row = b*1024 + n
      int b = g >> 10, n = g & 1023;
      float py = (float)pos[g * 2 + 0];
      float px = (float)pos[g * 2 + 1];
      float ay = py * fi, ax = px * fi;
      float cy = __cosf(ay), sy = __sinf(ay);
      float cx = __cosf(ax), sx = __sinf(ax);
      float y0 = acc0[v], y1 = acc1[v], x0 = acc2[v], x1 = acc3[v];
      size_t base = (((size_t)(b * NH + nt)) * SEQ + n) * HD + lmod;
      outb[base + 0]  = (bf16_t)(y0 * cy - y1 * sy);
      outb[base + 16] = (bf16_t)(y1 * cy + y0 * sy);
      outb[base + 32] = (bf16_t)(x0 * cx - x1 * sx);
      outb[base + 48] = (bf16_t)(x1 * cx + x0 * sx);
    }
  } else if (MODE == 1) {
#pragma unroll
    for (int v = 0; v < 8; ++v) {
      int g = mt * 16 + v + 8 * hi;
      int b = g >> 10, n = g & 1023;
      size_t hb = ((size_t)(b * NH + nt)) * HD;
      outb[(hb + 0 * 16 + lmod) * SEQ + n] = (bf16_t)acc0[v];
      outb[(hb + 1 * 16 + lmod) * SEQ + n] = (bf16_t)acc1[v];
      outb[(hb + 2 * 16 + lmod) * SEQ + n] = (bf16_t)acc2[v];
      outb[(hb + 3 * 16 + lmod) * SEQ + n] = (bf16_t)acc3[v];
    }
  } else {
    float b0 = bias[nt * 64 + 0 + lmod];
    float b1 = bias[nt * 64 + 16 + lmod];
    float b2 = bias[nt * 64 + 32 + lmod];
    float b3 = bias[nt * 64 + 48 + lmod];
#pragma unroll
    for (int v = 0; v < 8; ++v) {
      int g = mt * 16 + v + 8 * hi;
      float* op = outf + (size_t)g * CDIM + nt * 64 + lmod;
      op[0]  = acc0[v] + b0;
      op[16] = acc1[v] + b1;
      op[32] = acc2[v] + b2;
      op[48] = acc3[v] + b3;
    }
  }
}

// Flash attention. One workgroup = 8 waves = 8 q-tiles of one (b,h); K/V tiles
// are double-buffered in LDS via async global->LDS copies shared by all waves.
// S^T = K·Q^T trick: C-layout probability tiles are already WMMA A-fragments.
__global__ __launch_bounds__(32 * QW) void attn_kernel(const bf16_t* __restrict__ Qp,
                                                       const bf16_t* __restrict__ Kp,
                                                       const bf16_t* __restrict__ Vt,
                                                       bf16_t* __restrict__ X) {
  const int tid = threadIdx.x;
  const int w = tid >> 5;
  const int lane = tid & 31;
  const int lmod = lane & 15;
  const int hi = lane >> 4;
  const int kbA = hi ? 8 : 0;

  const int bh = blockIdx.x >> 3;  // 96 heads, SEQ/16/QW = 8 q-groups
  const int qg = blockIdx.x & 7;
  const int qt = qg * QW + w;
  const int b = bh / NH, h = bh % NH;

  const bf16_t* Qb = Qp + (size_t)bh * SEQ * HD;
  const bf16_t* Kb = Kp + (size_t)bh * SEQ * HD;
  const bf16_t* Vb = Vt + (size_t)bh * HD * SEQ;

  __shared__ __align__(32) bf16_t Kl[2][32 * HD];  // [buf][krow_local*64 + c]
  __shared__ __align__(32) bf16_t Vl[2][HD * 32];  // [buf][hd*32 + k_local]

  // Q as B operand of S^T: column n = qrow, K = head-dim chunk
  const bf16_t* qp = Qb + (size_t)(qt * 16 + lmod) * HD + kbA;
  const v16bf bq0 = ld2x8(qp);
  const v16bf bq1 = ld2x8(qp + 32);

  v8f o0 = {}, o1 = {}, o2 = {}, o3 = {};
  float mrun = -1e30f, lrun = 0.f;

  // Stage a 32-row K tile (contiguous 4KB) and V tile (64 rows x 64B) into LDS.
  auto stage = [&](int kt, int buf) {
    async_cp16(Kb + (size_t)kt * 32 * HD + tid * 8, &Kl[buf][tid * 8]);
    async_cp16(Vb + (size_t)(tid >> 2) * SEQ + kt * 32 + (tid & 3) * 8,
               &Vl[buf][(tid >> 2) * 32 + (tid & 3) * 8]);
  };

  stage(0, 0);
  wait_async0();
  __syncthreads();

  for (int kt = 0; kt < SEQ / 32; ++kt) {
    const int cur = kt & 1;
    if (kt + 1 < SEQ / 32) stage(kt + 1, cur ^ 1);  // overlaps with compute

    // S^T tiles for 32 k-rows x 16 q-rows (two 16x16 C tiles), K from LDS
    v8f st0, st1;
    {
      const bf16_t* kl0 = &Kl[cur][(size_t)lmod * HD + kbA];
      v8f z0 = {};
      st0 = wmma_bf16(ld2x8(kl0), bq0, z0);
      st0 = wmma_bf16(ld2x8(kl0 + 32), bq1, st0);
      const bf16_t* kl1 = kl0 + 16 * HD;
      v8f z1 = {};
      st1 = wmma_bf16(ld2x8(kl1), bq0, z1);
      st1 = wmma_bf16(ld2x8(kl1 + 32), bq1, st1);
    }
    // online softmax; this lane owns q-row lmod (k-index lives across VGPRs)
    float mloc = -1e30f;
#pragma unroll
    for (int v = 0; v < 8; ++v) {
      st0[v] *= 0.125f;  // 1/sqrt(64)
      st1[v] *= 0.125f;
      mloc = fmaxf(mloc, fmaxf(st0[v], st1[v]));
    }
    mloc = fmaxf(mloc, __shfl_xor(mloc, 16));
    const float mnew = fmaxf(mrun, mloc);
    const float corr = __expf(mrun - mnew);
    float p0[8], p1[8], rs = 0.f;
#pragma unroll
    for (int v = 0; v < 8; ++v) {
      p0[v] = __expf(st0[v] - mnew);
      p1[v] = __expf(st1[v] - mnew);
      rs += p0[v] + p1[v];
    }
    rs += __shfl_xor(rs, 16);
    lrun = lrun * corr + rs;
    mrun = mnew;

    // P in A-fragment layout for free (S^T trick)
    v16bf pa;
#pragma unroll
    for (int v = 0; v < 8; ++v) {
      pa[v] = (bf16_t)p0[v];
      pa[8 + v] = (bf16_t)p1[v];
    }
    // broadcast per-row correction into C/D row layout (row = v + 8*hi)
    float cv[8];
#pragma unroll
    for (int v = 0; v < 8; ++v) cv[v] = __shfl(corr, v + 8 * hi, 16);

    const bf16_t* vl = &Vl[cur][(size_t)lmod * 32 + kbA];
    {
      v8f c;
#pragma unroll
      for (int v = 0; v < 8; ++v) c[v] = o0[v] * cv[v];
      o0 = wmma_bf16(pa, ld2x8(vl + (size_t)0 * 16 * 32), c);
#pragma unroll
      for (int v = 0; v < 8; ++v) c[v] = o1[v] * cv[v];
      o1 = wmma_bf16(pa, ld2x8(vl + (size_t)1 * 16 * 32), c);
#pragma unroll
      for (int v = 0; v < 8; ++v) c[v] = o2[v] * cv[v];
      o2 = wmma_bf16(pa, ld2x8(vl + (size_t)2 * 16 * 32), c);
#pragma unroll
      for (int v = 0; v < 8; ++v) c[v] = o3[v] * cv[v];
      o3 = wmma_bf16(pa, ld2x8(vl + (size_t)3 * 16 * 32), c);
    }

    wait_async0();    // own next-tile copies done (ran under the WMMAs)
    __syncthreads();  // everyone done reading buf[cur] & staging buf[cur^1]
  }

  const float linv = 1.0f / lrun;
  float lv[8];
#pragma unroll
  for (int v = 0; v < 8; ++v) lv[v] = __shfl(linv, v + 8 * hi, 16);

#pragma unroll
  for (int v = 0; v < 8; ++v) {
    int grow = qt * 16 + v + 8 * hi;
    bf16_t* op = X + ((size_t)(b * SEQ + grow)) * CDIM + h * HD + lmod;
    op[0]  = (bf16_t)(o0[v] * lv[v]);
    op[16] = (bf16_t)(o1[v] * lv[v]);
    op[32] = (bf16_t)(o2[v] * lv[v]);
    op[48] = (bf16_t)(o3[v] * lv[v]);
  }
}

extern "C" void kernel_launch(void* const* d_in, const int* in_sizes, int n_in,
                              void* d_out, int out_size, void* d_ws, size_t ws_size,
                              hipStream_t stream) {
  const float* query = (const float*)d_in[0];
  const float* key_  = (const float*)d_in[1];
  const float* value = (const float*)d_in[2];
  const int*   qpos  = (const int*)d_in[3];
  const int*   kpos  = (const int*)d_in[4];
  const float* Wq    = (const float*)d_in[5];
  const float* Wk    = (const float*)d_in[6];
  const float* Wv    = (const float*)d_in[7];
  const float* Wo    = (const float*)d_in[8];
  const float* bo    = (const float*)d_in[9];
  float* out = (float*)d_out;

  const size_t per = (size_t)BATCH * NH * SEQ * HD;  // 6,291,456
  const size_t wsz = (size_t)CDIM * CDIM;            // 589,824
  bf16_t* Qp  = (bf16_t*)d_ws;   // roped Q   [b][h][n][hd]
  bf16_t* Kp  = Qp + per;        // roped K   [b][h][n][hd]
  bf16_t* Vt  = Kp + per;        // V         [b][h][hd][n]
  bf16_t* Xa  = Vt + per;        // attn out  [b][n][c]
  bf16_t* qb  = Xa + per;        // bf16 activations
  bf16_t* kb  = qb + per;
  bf16_t* vb  = kb + per;
  bf16_t* wq16 = vb + per;       // bf16 weights
  bf16_t* wk16 = wq16 + wsz;
  bf16_t* wv16 = wk16 + wsz;
  bf16_t* wo16 = wv16 + wsz;

  // one-pass fp32 -> bf16 conversions
  const int a4 = (int)(per / 4), w4 = (int)(wsz / 4);
  cvt_kernel<<<dim3((a4 + 255) / 256), dim3(256), 0, stream>>>(query, qb, a4);
  cvt_kernel<<<dim3((a4 + 255) / 256), dim3(256), 0, stream>>>(key_,  kb, a4);
  cvt_kernel<<<dim3((a4 + 255) / 256), dim3(256), 0, stream>>>(value, vb, a4);
  cvt_kernel<<<dim3((w4 + 255) / 256), dim3(256), 0, stream>>>(Wq, wq16, w4);
  cvt_kernel<<<dim3((w4 + 255) / 256), dim3(256), 0, stream>>>(Wk, wk16, w4);
  cvt_kernel<<<dim3((w4 + 255) / 256), dim3(256), 0, stream>>>(Wv, wv16, w4);
  cvt_kernel<<<dim3((w4 + 255) / 256), dim3(256), 0, stream>>>(Wo, wo16, w4);

  dim3 grid(MT * NT), blk(32);
  proj_kernel<0><<<grid, blk, 0, stream>>>(qb, wq16, qpos, nullptr, Qp, nullptr);
  proj_kernel<0><<<grid, blk, 0, stream>>>(kb, wk16, kpos, nullptr, Kp, nullptr);
  proj_kernel<1><<<grid, blk, 0, stream>>>(vb, wv16, nullptr, nullptr, Vt, nullptr);
  attn_kernel<<<dim3(BATCH * NH * (SEQ / 16 / QW)), dim3(32 * QW), 0, stream>>>(Qp, Kp, Vt, Xa);
  proj_kernel<2><<<grid, blk, 0, stream>>>(Xa, wo16, nullptr, bo, nullptr, out);
}
